// LinearAttention_82910048682080
// MI455X (gfx1250) — compile-verified, matrix operand order checked by reading, not run on hardware
//
#include <hip/hip_runtime.h>
#include <hip/hip_bf16.h>

// MI455X / gfx1250: all four GEMMs via v_wmma_f32_16x16x32_bf16, async
// global->LDS staging pipelined on ASYNCcnt, split-K kv-state with fp32
// global atomics. Shapes hardcoded: B=16, N=4096, C=768, H=12, d=64.

typedef __bf16 bf16_t;
typedef __attribute__((ext_vector_type(16))) __bf16 v16bf;
typedef __attribute__((ext_vector_type(8)))  __bf16 v8bf;
typedef __attribute__((ext_vector_type(8)))  float  v8f;
typedef __attribute__((ext_vector_type(4)))  float  v4f;

#define USE_ASYNC_LDS 1

__device__ __forceinline__ v8f wmma_bf16(v16bf a, v16bf b, v8f c) {
  return __builtin_amdgcn_wmma_f32_16x16x32_bf16(false, a, false, b, (short)0, c,
                                                 false, false);
}

// 16-byte global -> LDS copy. Async form uses GLOBAL_LOAD_ASYNC_TO_LDS_B128
// (ISA 15.18.3 op 98, tracked with ASYNCcnt). VDST operand is the LDS byte
// address = low 32 bits of the flat pointer (ISA 10.2).
__device__ __forceinline__ void copy16_g2l(const bf16_t* __restrict__ g, bf16_t* l) {
#if USE_ASYNC_LDS
  unsigned loff = (unsigned)(size_t)l;
  unsigned long long ga = (unsigned long long)(size_t)g;
  asm volatile("global_load_async_to_lds_b128 %0, %1, off"
               :: "v"(loff), "v"(ga) : "memory");
#else
  *(v8bf*)l = *(const v8bf*)g;
#endif
}

__device__ __forceinline__ void wait_async_keep2() {
#if USE_ASYNC_LDS
  asm volatile("s_wait_asynccnt 0x2" ::: "memory");   // drain current tile, keep next in flight
#endif
}
__device__ __forceinline__ void wait_async_all() {
#if USE_ASYNC_LDS
  asm volatile("s_wait_asynccnt 0x0" ::: "memory");
#endif
}

// Stage one 128x32 bf16 tile (row stride 768 in global) into LDS (stride 40).
// 512 x 16B chunks, 2 per thread.
__device__ __forceinline__ void stage_tile(const bf16_t* __restrict__ gbase,
                                           bf16_t* sbuf, int tid) {
  #pragma unroll
  for (int it = 0; it < 2; ++it) {
    int idx = tid + it * 256;
    int r = idx >> 2;
    int c8 = (idx & 3) * 8;
    copy16_g2l(gbase + (long)r * 768 + c8, &sbuf[r * 40 + c8]);
  }
}

// WMMA A/B bf16 fragment from a row-major matrix (ISA 7.12.2 16-bit 16x32
// layout): lanes 0-15 hold K chunks [0..7],[16..23]; lanes 16-31 [8..15],[24..31].
__device__ __forceinline__ v16bf load_frag(const bf16_t* __restrict__ base,
                                           long stride, int row, int k0, int lane) {
  const int c0 = (lane & 16) ? 8 : 0;
  const bf16_t* p = base + (long)row * stride + (k0 + c0);
  v16bf r;
  ((v8bf*)&r)[0] = *(const v8bf*)(p);
  ((v8bf*)&r)[1] = *(const v8bf*)(p + 16);
  return r;
}

// ---------------------------------------------------------------- convert
__global__ __launch_bounds__(256)
void convert_f32_bf16(const float* __restrict__ in, bf16_t* __restrict__ out, int n4) {
  int i = blockIdx.x * 256 + threadIdx.x;
  if (i < n4) {
    v4f f = *(const v4f*)(in + (long)i * 4);
    bf16_t* d = out + (long)i * 4;
    d[0] = (bf16_t)f[0]; d[1] = (bf16_t)f[1]; d[2] = (bf16_t)f[2]; d[3] = (bf16_t)f[3];
  }
}

// ---------------------------------------------------------------- GEMM 1: qkv
// C[65536,2304] = Xbf[65536,768] @ Wqkv^T, fused relu + scatter into
// Q [B,H,N,d], KT [B,H,d,N], VT [B,H,d,N]. Xbf (100MB) fits in the 192MB L2,
// so the 18x column-block re-reads stay on-chip.
__global__ __launch_bounds__(256)
void qkv_gemm(const bf16_t* __restrict__ Xb, const bf16_t* __restrict__ Wb,
              bf16_t* __restrict__ Qb, bf16_t* __restrict__ KT,
              bf16_t* __restrict__ VT) {
  __shared__ bf16_t sA[2][128 * 40];
  const int tid = threadIdx.x;
  const int lane = tid & 31;
  const int wave = tid >> 5;
  const int waveM = wave >> 2;      // 0..1
  const int waveN = wave & 3;       // 0..3
  const int n0 = blockIdx.x * 128;
  const long m0 = (long)blockIdx.y * 128;
  const bf16_t* src = Xb + m0 * 768;

  v8f z = {};
  v8f acc[4][2];
  #pragma unroll
  for (int i = 0; i < 4; ++i)
    #pragma unroll
    for (int j = 0; j < 2; ++j) acc[i][j] = z;

  stage_tile(src, sA[0], tid);
  int buf = 0;
  for (int k0 = 0; k0 < 768; k0 += 32, buf ^= 1) {
    if (k0 + 32 < 768) {
      stage_tile(src + k0 + 32, sA[buf ^ 1], tid);
      wait_async_keep2();
    } else {
      wait_async_all();
    }
    __syncthreads();
    const bf16_t* cur = sA[buf];

    v16bf bfrag[2];
    #pragma unroll
    for (int tn = 0; tn < 2; ++tn)
      bfrag[tn] = load_frag(Wb, 768, n0 + waveN * 32 + tn * 16 + (lane & 15), k0, lane);
    #pragma unroll
    for (int tm = 0; tm < 4; ++tm) {
      v16bf afrag = load_frag(cur, 40, waveM * 64 + tm * 16 + (lane & 15), 0, lane);
      #pragma unroll
      for (int tn = 0; tn < 2; ++tn)
        acc[tm][tn] = wmma_bf16(afrag, bfrag[tn], acc[tm][tn]);
    }
    __syncthreads();
  }

  // D layout: VGPR r, lanes 0-15 -> M=r, lanes 16-31 -> M=r+8; N = lane&15.
  #pragma unroll
  for (int tm = 0; tm < 4; ++tm)
    #pragma unroll
    for (int tn = 0; tn < 2; ++tn)
      #pragma unroll
      for (int r = 0; r < 8; ++r) {
        long m = m0 + waveM * 64 + tm * 16 + r + ((lane >> 4) << 3);
        int j = n0 + waveN * 32 + tn * 16 + (lane & 15);
        float val = acc[tm][tn][r];
        int which = j / 768;            // 0=q 1=k 2=v
        int c = j - which * 768;
        int h = c >> 6, dd = c & 63;
        int b = (int)(m >> 12);
        int tok = (int)(m & 4095);
        long bh = b * 12 + h;
        if (which == 0) {
          Qb[(bh * 4096 + tok) * 64 + dd] = (bf16_t)(val > 0.f ? val : 0.f);
        } else if (which == 1) {
          KT[(bh * 64 + dd) * 4096 + tok] = (bf16_t)(val > 0.f ? val : 0.f);
        } else {
          VT[(bh * 64 + dd) * 4096 + tok] = (bf16_t)val;
        }
      }
}

// ---------------------------------------------------------------- ksum / denom
__global__ __launch_bounds__(64)
void ksum_kernel(const bf16_t* __restrict__ KT, float* __restrict__ Ksum) {
  int bh = blockIdx.x;          // 192
  int dd = threadIdx.x;         // 64
  const bf16_t* row = KT + ((long)bh * 64 + dd) * 4096;
  float s = 0.f;
  for (int n = 0; n < 4096; n += 8) {
    v8bf v = *(const v8bf*)(row + n);
    #pragma unroll
    for (int i = 0; i < 8; ++i) s += (float)v[i];
  }
  Ksum[bh * 64 + dd] = s;
}

__global__ __launch_bounds__(256)
void denom_kernel(const bf16_t* __restrict__ Qb, const float* __restrict__ Ksum,
                  float* __restrict__ Denom) {
  long i = (long)blockIdx.x * 256 + threadIdx.x;   // over B*H*N = 786432
  int bh = (int)(i >> 12);
  const bf16_t* q = Qb + i * 64;
  const float* ks = Ksum + (long)bh * 64;
  float s = 0.f;
  #pragma unroll
  for (int d8 = 0; d8 < 64; d8 += 8) {
    v8bf v = *(const v8bf*)(q + d8);
    #pragma unroll
    for (int j = 0; j < 8; ++j) s += (float)v[j] * ks[d8 + j];
  }
  Denom[i] = fmaxf(s, 100.0f);
}

// ---------------------------------------------------------------- GEMM 2: kv state
// Split-K (8 ways): per (split, bh) accumulate a partial kv[64,64] over 512
// K-values with WMMA, then global_atomic_add_f32 into the fp32 accumulator.
__global__ __launch_bounds__(256)
void kv_gemm_splitk(const bf16_t* __restrict__ KT, const bf16_t* __restrict__ VT,
                    float* __restrict__ KVF) {
  int split = blockIdx.x;         // 0..7
  int bh = blockIdx.y;            // 0..191
  int lane = threadIdx.x & 31;
  int wave = threadIdx.x >> 5;
  int tileM = wave >> 1;          // 0..3
  int tnb = (wave & 1) * 2;       // col tiles {tnb, tnb+1}
  const bf16_t* A = KT + (long)bh * 64 * 4096;
  const bf16_t* Bm = VT + (long)bh * 64 * 4096;
  v8f z = {};
  v8f acc[2] = {z, z};
  int nbeg = split * 512;
  for (int n0 = nbeg; n0 < nbeg + 512; n0 += 32) {
    __builtin_prefetch(A + (long)(tileM * 16 + (lane & 15)) * 4096 + n0 + 128, 0, 1);
    v16bf a = load_frag(A, 4096, tileM * 16 + (lane & 15), n0, lane);
    #pragma unroll
    for (int t = 0; t < 2; ++t) {
      v16bf b = load_frag(Bm, 4096, (tnb + t) * 16 + (lane & 15), n0, lane);
      acc[t] = wmma_bf16(a, b, acc[t]);
    }
  }
  #pragma unroll
  for (int t = 0; t < 2; ++t)
    #pragma unroll
    for (int r = 0; r < 8; ++r) {
      int dd = tileM * 16 + r + ((lane >> 4) << 3);
      int e  = (tnb + t) * 16 + (lane & 15);
      unsafeAtomicAdd(&KVF[((long)bh * 64 + dd) * 64 + e], acc[t][r]);
    }
}

// Apply temperature, write kv transposed (kvT[e][dd]) in bf16 for GEMM3.
__global__ __launch_bounds__(256)
void kv_finalize(const float* __restrict__ KVF, const float* __restrict__ Temp,
                 bf16_t* __restrict__ KVT) {
  int bh = blockIdx.x;
  float t = Temp[bh % 12];
  for (int i = threadIdx.x; i < 4096; i += 256) {
    int dd = i >> 6, e = i & 63;
    KVT[((long)bh * 64 + e) * 64 + dd] = (bf16_t)(KVF[(long)bh * 4096 + i] * t);
  }
}

// ---------------------------------------------------------------- GEMM 3: q @ kv
__global__ __launch_bounds__(256)
void attn_out_gemm(const bf16_t* __restrict__ Qb, const bf16_t* __restrict__ KVT,
                   const float* __restrict__ Denom, bf16_t* __restrict__ Y) {
  int bh = blockIdx.y;            // 192
  int b = bh / 12, h = bh % 12;
  int m0 = blockIdx.x * 128;      // token block
  int lane = threadIdx.x & 31;
  int wave = threadIdx.x >> 5;    // owns 16 rows
  const bf16_t* A = Qb + (long)bh * 4096 * 64;
  const bf16_t* Bm = KVT + (long)bh * 64 * 64;
  v8f z = {};
  v8f acc[4] = {z, z, z, z};
  #pragma unroll
  for (int k0 = 0; k0 < 64; k0 += 32) {
    v16bf a = load_frag(A, 64, m0 + wave * 16 + (lane & 15), k0, lane);
    #pragma unroll
    for (int tn = 0; tn < 4; ++tn) {
      v16bf bb = load_frag(Bm, 64, tn * 16 + (lane & 15), k0, lane);
      acc[tn] = wmma_bf16(a, bb, acc[tn]);
    }
  }
  #pragma unroll
  for (int tn = 0; tn < 4; ++tn)
    #pragma unroll
    for (int r = 0; r < 8; ++r) {
      int tok = m0 + wave * 16 + r + ((lane >> 4) << 3);
      int e = tn * 16 + (lane & 15);
      float val = acc[tn][r] / Denom[(long)bh * 4096 + tok];
      Y[((long)(b * 4096 + tok)) * 768 + h * 64 + e] = (bf16_t)val;
    }
}

// ---------------------------------------------------------------- GEMM 4: output proj
__global__ __launch_bounds__(256)
void proj_gemm(const bf16_t* __restrict__ Y, const bf16_t* __restrict__ Wb,
               const float* __restrict__ bias, float* __restrict__ Out) {
  __shared__ bf16_t sA[2][128 * 40];
  int tid = threadIdx.x, lane = tid & 31, wave = tid >> 5;
  int waveM = wave >> 2, waveN = wave & 3;
  int n0 = blockIdx.x * 128;
  long m0 = (long)blockIdx.y * 128;
  const bf16_t* src = Y + m0 * 768;
  v8f z = {};
  v8f acc[4][2];
  #pragma unroll
  for (int i = 0; i < 4; ++i)
    #pragma unroll
    for (int j = 0; j < 2; ++j) acc[i][j] = z;

  stage_tile(src, sA[0], tid);
  int buf = 0;
  for (int k0 = 0; k0 < 768; k0 += 32, buf ^= 1) {
    if (k0 + 32 < 768) {
      stage_tile(src + k0 + 32, sA[buf ^ 1], tid);
      wait_async_keep2();
    } else {
      wait_async_all();
    }
    __syncthreads();
    const bf16_t* cur = sA[buf];
    v16bf bfrag[2];
    #pragma unroll
    for (int tn = 0; tn < 2; ++tn)
      bfrag[tn] = load_frag(Wb, 768, n0 + waveN * 32 + tn * 16 + (lane & 15), k0, lane);
    #pragma unroll
    for (int tm = 0; tm < 4; ++tm) {
      v16bf a = load_frag(cur, 40, waveM * 64 + tm * 16 + (lane & 15), 0, lane);
      #pragma unroll
      for (int tn = 0; tn < 2; ++tn)
        acc[tm][tn] = wmma_bf16(a, bfrag[tn], acc[tm][tn]);
    }
    __syncthreads();
  }
  #pragma unroll
  for (int tm = 0; tm < 4; ++tm)
    #pragma unroll
    for (int tn = 0; tn < 2; ++tn)
      #pragma unroll
      for (int r = 0; r < 8; ++r) {
        long m = m0 + waveM * 64 + tm * 16 + r + ((lane >> 4) << 3);
        int j = n0 + waveN * 32 + tn * 16 + (lane & 15);
        Out[m * 768 + j] = acc[tm][tn][r] + bias[j];
      }
}

// ---------------------------------------------------------------- launch
extern "C" void kernel_launch(void* const* d_in, const int* in_sizes, int n_in,
                              void* d_out, int out_size, void* d_ws, size_t ws_size,
                              hipStream_t stream) {
  const float* x      = (const float*)d_in[0];   // [16,4096,768]
  const float* qkv_w  = (const float*)d_in[1];   // [2304,768]
  const float* proj_w = (const float*)d_in[2];   // [768,768]
  const float* proj_b = (const float*)d_in[3];   // [768]
  const float* temp   = (const float*)d_in[4];   // [12,1,1]
  (void)in_sizes; (void)n_in; (void)out_size; (void)ws_size;

  char* ws = (char*)d_ws;
  size_t off = 0;
  auto alloc = [&](size_t bytes) -> char* {
    char* p = ws + off;
    off = (off + bytes + 255) & ~(size_t)255;
    return p;
  };
  bf16_t* xbf   = (bf16_t*)alloc(65536ull * 768 * 2);      // bf16 x (L2-resident)
  bf16_t* wqkv  = (bf16_t*)alloc(2304ull * 768 * 2);
  bf16_t* wproj = (bf16_t*)alloc(768ull * 768 * 2);
  bf16_t* Qb    = (bf16_t*)alloc(192ull * 4096 * 64 * 2);  // [B,H,N,d]
  bf16_t* KT    = (bf16_t*)alloc(192ull * 64 * 4096 * 2);  // [B,H,d,N]
  bf16_t* VT    = (bf16_t*)alloc(192ull * 64 * 4096 * 2);  // [B,H,d,N]
  float*  KVF   = (float*)alloc(192ull * 64 * 64 * 4);     // fp32 split-K accum
  bf16_t* KVT   = (bf16_t*)alloc(192ull * 64 * 64 * 2);    // [B,H,e,d]
  float*  Ksum  = (float*)alloc(192ull * 64 * 4);
  float*  Denom = (float*)alloc(192ull * 4096 * 4);
  bf16_t* Y     = (bf16_t*)alloc(65536ull * 768 * 2);      // [B,N,C]
  float*  Out   = (float*)d_out;

  convert_f32_bf16<<<65536 * 768 / 4 / 256, 256, 0, stream>>>(x, xbf, 65536 * 768 / 4);
  convert_f32_bf16<<<2304 * 768 / 4 / 256, 256, 0, stream>>>(qkv_w, wqkv, 2304 * 768 / 4);
  convert_f32_bf16<<<768 * 768 / 4 / 256, 256, 0, stream>>>(proj_w, wproj, 768 * 768 / 4);

  qkv_gemm<<<dim3(18, 512), 256, 0, stream>>>(xbf, wqkv, Qb, KT, VT);
  ksum_kernel<<<192, 64, 0, stream>>>(KT, Ksum);
  denom_kernel<<<192 * 4096 / 256, 256, 0, stream>>>(Qb, Ksum, Denom);

  hipMemsetAsync(KVF, 0, 192ull * 64 * 64 * 4, stream);
  kv_gemm_splitk<<<dim3(8, 192), 256, 0, stream>>>(KT, VT, KVF);
  kv_finalize<<<192, 256, 0, stream>>>(KVF, temp, KVT);

  attn_out_gemm<<<dim3(32, 192), 256, 0, stream>>>(Qb, KVT, Denom, Y);
  proj_gemm<<<dim3(6, 512), 256, 0, stream>>>(Y, wproj, proj_b, Out);
}
